// MultiHeadAttention_867583394552
// MI455X (gfx1250) — matrix operand order repaired and checked by reading, not compiled, not run on hardware
//
#include <hip/hip_runtime.h>

// ---------------------------------------------------------------------------
// MI455X (gfx1250) multi-head attention: WMMA bf16 GEMMs + flash attention.
// B=4, S=2048, D=1024, H=16, d_k=64.  Wave32, 16x16x32 bf16 WMMA, f32 accum.
// Workspace usage: 4 * 16 MiB = 64 MiB bf16 intermediates.
// ---------------------------------------------------------------------------

typedef __bf16 bf16_t;
typedef __attribute__((ext_vector_type(8)))  __bf16 v8bf;
typedef __attribute__((ext_vector_type(16))) __bf16 v16bf;
typedef __attribute__((ext_vector_type(8)))  float  v8f;

#define D_MODEL 1024
#define SEQ     2048
#define NHEAD   16
#define DK      64

__device__ __forceinline__ bf16_t f2bf(float f) {
  unsigned u = __builtin_bit_cast(unsigned, f);
  u += 0x7FFFu + ((u >> 16) & 1u);           // round-to-nearest-even
  unsigned short h = (unsigned short)(u >> 16);
  return __builtin_bit_cast(bf16_t, h);
}

__device__ __forceinline__ v16bf cat8(v8bf lo, v8bf hi) {
  return __builtin_shufflevector(lo, hi, 0,1,2,3,4,5,6,7,8,9,10,11,12,13,14,15);
}

// xor-shuffle within groups of 16 lanes via ds_swizzle (and=0x1F, or=0, xor=XM)
template <int XM>
__device__ __forceinline__ float swz_xor(float v) {
  int i = __builtin_bit_cast(int, v);
  i = __builtin_amdgcn_ds_swizzle(i, (XM << 10) | 0x1F);
  return __builtin_bit_cast(float, i);
}
__device__ __forceinline__ float redmax16(float v) {
  v = fmaxf(v, swz_xor<1>(v)); v = fmaxf(v, swz_xor<2>(v));
  v = fmaxf(v, swz_xor<4>(v)); v = fmaxf(v, swz_xor<8>(v));
  return v;
}
__device__ __forceinline__ float redsum16(float v) {
  v += swz_xor<1>(v); v += swz_xor<2>(v);
  v += swz_xor<4>(v); v += swz_xor<8>(v);
  return v;
}

// Convert 16 consecutive fp32 -> 16 bf16 and store to LDS (two b128 stores).
__device__ __forceinline__ void stage16_f32(const float* __restrict__ src, bf16_t* dst) {
  float4 f0 = ((const float4*)src)[0];
  float4 f1 = ((const float4*)src)[1];
  float4 f2 = ((const float4*)src)[2];
  float4 f3 = ((const float4*)src)[3];
  v8bf p0, p1;
  p0[0]=f2bf(f0.x); p0[1]=f2bf(f0.y); p0[2]=f2bf(f0.z); p0[3]=f2bf(f0.w);
  p0[4]=f2bf(f1.x); p0[5]=f2bf(f1.y); p0[6]=f2bf(f1.z); p0[7]=f2bf(f1.w);
  p1[0]=f2bf(f2.x); p1[1]=f2bf(f2.y); p1[2]=f2bf(f2.z); p1[3]=f2bf(f2.w);
  p1[4]=f2bf(f3.x); p1[5]=f2bf(f3.y); p1[6]=f2bf(f3.z); p1[7]=f2bf(f3.w);
  *(v8bf*)dst       = p0;
  *(v8bf*)(dst + 8) = p1;
}
__device__ __forceinline__ void stage32_f32(const float* __restrict__ src, bf16_t* dst) {
  stage16_f32(src, dst);
  stage16_f32(src + 16, dst + 16);
}

// ---------------------------------------------------------------------------
// GEMM: out[m,n] = sum_k A[m,k] * W[n,k] + bias[n]    (torch Linear: x @ W^T)
// M = 8192 (B*S), N = K = 1024. Block tile 128x128, K-step 64 (2 WMMA steps).
// 8 waves: 4 (M, 32 rows each) x 2 (N, 64 cols each); per wave 2x4 WMMA tiles.
// MODE 0: bf16 out, head-split [B,H,S,64]            (Q, K projections)
// MODE 1: bf16 out, head-split transposed [B,H,64,S] (V projection)
// MODE 2: fp32 out, row-major [M,1024]; bf16 A staged via async-to-LDS.
// ---------------------------------------------------------------------------
template <typename AT, int MODE>
__global__ __launch_bounds__(256) void mha_gemm(const AT* __restrict__ A,
                                                const float* __restrict__ W,
                                                const float* __restrict__ bias,
                                                void* __restrict__ outp) {
  constexpr int K = D_MODEL;
  __shared__ bf16_t sA[128][72];   // 144B row stride: 16B aligned, conflict-free
  __shared__ bf16_t sB[128][72];   // sB[n][k]: columns of B == rows of W

  const int tid  = threadIdx.x;
  const int lane = tid & 31;
  const int wave = tid >> 5;
  const int hf   = lane >> 4;      // 0/1 : K-half selector in fragments
  const int ln   = lane & 15;      // row (A) / col (B,C,D) index within tile
  const int wm   = wave & 3;
  const int wn   = wave >> 2;
  const int rowBase = blockIdx.x * 128;
  const int colBase = blockIdx.y * 128;

  const int lr = tid >> 1;         // 0..127 tile row staged by this thread
  const int lc = (tid & 1) << 5;   // 0 / 32  (32 elements per thread)

  v8f acc[2][4];
#pragma unroll
  for (int i = 0; i < 2; ++i)
#pragma unroll
    for (int j = 0; j < 4; ++j)
#pragma unroll
      for (int r = 0; r < 8; ++r) acc[i][j][r] = 0.f;

  const float* wrow = W + (size_t)(colBase + lr) * K + lc;

  for (int k0 = 0; k0 < K; k0 += 64) {
    // ---- stage A tile ----
    if constexpr (sizeof(AT) == 4) {
      const float* ap = (const float*)A + (size_t)(rowBase + lr) * K + k0 + lc;
      stage32_f32(ap, &sA[lr][lc]);
      if (k0 + 64 < K) __builtin_prefetch(ap + 64, 0, 0);
    } else {
      // bf16 A: pure byte copy -> CDNA5 async global->LDS DMA (ASYNCcnt).
      // ISA: INST_OFFSET is added to BOTH the LDS and the global address,
      // so one base pair + immediate offsets copies 64 contiguous bytes.
      const bf16_t* ap = (const bf16_t*)A + (size_t)(rowBase + lr) * K + k0 + lc;
      unsigned ldsoff = (unsigned)(size_t)&sA[lr][lc];
      asm volatile(
          "global_load_async_to_lds_b128 %0, %1, off\n\t"
          "global_load_async_to_lds_b128 %0, %1, off offset:16\n\t"
          "global_load_async_to_lds_b128 %0, %1, off offset:32\n\t"
          "global_load_async_to_lds_b128 %0, %1, off offset:48"
          :: "v"(ldsoff), "v"(ap) : "memory");
      if (k0 + 64 < K) __builtin_prefetch(ap + 64, 0, 0);
    }
    // ---- stage B tile (fp32 weights -> bf16) ----
    {
      const float* wp = wrow + k0;
      stage32_f32(wp, &sB[lr][lc]);
      if (k0 + 64 < K) __builtin_prefetch(wp + 64, 0, 0);
    }
    if constexpr (sizeof(AT) == 2) {
      asm volatile("s_wait_asynccnt 0x0" ::: "memory");  // LDS writes landed
    }
    __syncthreads();

#pragma unroll
    for (int kk = 0; kk < 2; ++kk) {
      // A fragment: lane = row, two 8-half chunks at K = hf*8 and 16+hf*8
      v16bf afrag[2];
#pragma unroll
      for (int i = 0; i < 2; ++i) {
        const bf16_t* ar = &sA[wm * 32 + i * 16 + ln][kk * 32];
        afrag[i] = cat8(*(const v8bf*)(ar + hf * 8),
                        *(const v8bf*)(ar + 16 + hf * 8));
      }
      // B fragment: lane = column, two 8-half chunks at K = hf*8 and 16+hf*8
#pragma unroll
      for (int j = 0; j < 4; ++j) {
        const bf16_t* br = &sB[wn * 64 + j * 16 + ln][kk * 32];
        v16bf bfrag = cat8(*(const v8bf*)(br + hf * 8),
                           *(const v8bf*)(br + 16 + hf * 8));
#pragma unroll
        for (int i = 0; i < 2; ++i)
          acc[i][j] = __builtin_amdgcn_wmma_f32_16x16x32_bf16(
              false, afrag[i], false, bfrag, (short)0, acc[i][j], false, false);
      }
    }
    __syncthreads();
  }

  // ---- epilogue: C/D layout is (m = r + 8*hf, n = ln) ----
#pragma unroll
  for (int j = 0; j < 4; ++j) {
    const int n = colBase + wn * 64 + j * 16 + ln;
    const float bv = bias[n];
#pragma unroll
    for (int i = 0; i < 2; ++i) {
#pragma unroll
      for (int r = 0; r < 8; ++r) {
        const int m = rowBase + wm * 32 + i * 16 + r + 8 * hf;
        const float val = acc[i][j][r] + bv;
        if constexpr (MODE == 0) {
          const int bb = m >> 11, ss = m & 2047, hh = n >> 6, dd = n & 63;
          ((bf16_t*)outp)[(((size_t)bb * NHEAD + hh) * SEQ + ss) * DK + dd] = f2bf(val);
        } else if constexpr (MODE == 1) {
          const int bb = m >> 11, ss = m & 2047, hh = n >> 6, dd = n & 63;
          ((bf16_t*)outp)[(((size_t)bb * NHEAD + hh) * DK + dd) * SEQ + ss] = f2bf(val);
        } else {
          ((float*)outp)[(size_t)m * D_MODEL + n] = val;
        }
      }
    }
  }
}

// ---------------------------------------------------------------------------
// Flash attention (causal), per (b,h): Q[S,64] x K[S,64]^T -> online softmax
// -> x V.  Block = 4 waves, 64 query rows; streams 64-key blocks.
// Q,K bf16 [B,H,S,64]; V bf16 transposed [B,H,64,S]; out bf16 [B,S,1024].
// ---------------------------------------------------------------------------
__global__ __launch_bounds__(128) void mha_flash(const bf16_t* __restrict__ Q,
                                                 const bf16_t* __restrict__ Kh,
                                                 const bf16_t* __restrict__ Vt,
                                                 bf16_t* __restrict__ Aout) {
  const int bh = blockIdx.y;           // b*16 + h
  const int b  = bh >> 4, h = bh & 15;
  const int qb = blockIdx.x;
  const int q0 = qb * 64;
  const int tid  = threadIdx.x;
  const int wave = tid >> 5, lane = tid & 31;
  const int hf = lane >> 4, ln = lane & 15;

  const bf16_t* Qb_ = Q  + (size_t)bh * SEQ * DK;
  const bf16_t* Kb_ = Kh + (size_t)bh * SEQ * DK;
  const bf16_t* Vb_ = Vt + (size_t)bh * DK * SEQ;

  __shared__ bf16_t sP[4][16][72];     // per-wave P tile, 144B row stride

  // Q A-fragments (lane = query row), d split into two K=32 steps
  v16bf qf[2];
  {
    const bf16_t* qp = Qb_ + (size_t)(q0 + wave * 16 + ln) * DK;
#pragma unroll
    for (int kk = 0; kk < 2; ++kk)
      qf[kk] = cat8(*(const v8bf*)(qp + kk * 32 + hf * 8),
                    *(const v8bf*)(qp + kk * 32 + 16 + hf * 8));
  }

  v8f of[4];
  float mrow[8], lrow[8];
#pragma unroll
  for (int nt = 0; nt < 4; ++nt)
#pragma unroll
    for (int r = 0; r < 8; ++r) of[nt][r] = 0.f;
#pragma unroll
  for (int r = 0; r < 8; ++r) { mrow[r] = -1e30f; lrow[r] = 0.f; }

  for (int jb = 0; jb <= qb; ++jb) {
    const int j0 = jb * 64;

    // ---- S = Q K^T : B columns = K rows (contiguous) ----
    v8f sf[4];
#pragma unroll
    for (int nt = 0; nt < 4; ++nt)
#pragma unroll
      for (int r = 0; r < 8; ++r) sf[nt][r] = 0.f;
#pragma unroll
    for (int kk = 0; kk < 2; ++kk) {
#pragma unroll
      for (int nt = 0; nt < 4; ++nt) {
        const bf16_t* kp = Kb_ + (size_t)(j0 + nt * 16 + ln) * DK + kk * 32 + hf * 8;
        v16bf kf = cat8(((const v8bf*)kp)[0], *(const v8bf*)(kp + 16));
        sf[nt] = __builtin_amdgcn_wmma_f32_16x16x32_bf16(
            false, qf[kk], false, kf, (short)0, sf[nt], false, false);
      }
    }

    // ---- scale, causal mask, online softmax ----
    const bool diag = (jb == qb);
    float rmax[8];
#pragma unroll
    for (int r = 0; r < 8; ++r) rmax[r] = -1e30f;
#pragma unroll
    for (int nt = 0; nt < 4; ++nt) {
#pragma unroll
      for (int r = 0; r < 8; ++r) {
        float v = sf[nt][r] * 0.125f;                     // 1/sqrt(64)
        if (diag && (j0 + nt * 16 + ln) > (q0 + wave * 16 + r + 8 * hf))
          v = -1e30f;
        sf[nt][r] = v;
        rmax[r] = fmaxf(rmax[r], v);
      }
    }
#pragma unroll
    for (int r = 0; r < 8; ++r) rmax[r] = redmax16(rmax[r]);

    float alpha[8], rsum[8];
#pragma unroll
    for (int r = 0; r < 8; ++r) {
      const float nm = fmaxf(mrow[r], rmax[r]);
      alpha[r] = __expf(mrow[r] - nm);
      mrow[r] = nm;
      rsum[r] = 0.f;
    }
#pragma unroll
    for (int nt = 0; nt < 4; ++nt) {
#pragma unroll
      for (int r = 0; r < 8; ++r) {
        const float p = __expf(sf[nt][r] - mrow[r]);
        sf[nt][r] = p;
        rsum[r] += p;
      }
    }
#pragma unroll
    for (int r = 0; r < 8; ++r) rsum[r] = redsum16(rsum[r]);
#pragma unroll
    for (int r = 0; r < 8; ++r) lrow[r] = lrow[r] * alpha[r] + rsum[r];
#pragma unroll
    for (int nt = 0; nt < 4; ++nt)
#pragma unroll
      for (int r = 0; r < 8; ++r) of[nt][r] *= alpha[r];

    // ---- transpose P: C-layout -> A-layout via per-wave LDS ----
#pragma unroll
    for (int nt = 0; nt < 4; ++nt)
#pragma unroll
      for (int r = 0; r < 8; ++r)
        sP[wave][r + 8 * hf][nt * 16 + ln] = f2bf(sf[nt][r]);
    asm volatile("s_wait_dscnt 0x0" ::: "memory");   // same-wave DS in-order

    // ---- O += P V : B columns of V are rows of Vt (contiguous) ----
#pragma unroll
    for (int kk = 0; kk < 2; ++kk) {
      const bf16_t* pp = &sP[wave][ln][kk * 32];
      v16bf pf = cat8(*(const v8bf*)(pp + hf * 8),
                      *(const v8bf*)(pp + 16 + hf * 8));
#pragma unroll
      for (int nt = 0; nt < 4; ++nt) {
        const bf16_t* vp = Vb_ + (size_t)(nt * 16 + ln) * SEQ + j0 + kk * 32 + hf * 8;
        v16bf vf = cat8(((const v8bf*)vp)[0], *(const v8bf*)(vp + 16));
        of[nt] = __builtin_amdgcn_wmma_f32_16x16x32_bf16(
            false, pf, false, vf, (short)0, of[nt], false, false);
      }
    }
    asm volatile("s_wait_dscnt 0x0" ::: "memory");
  }

  // ---- normalize and write [B,S,H*64] bf16 ----
#pragma unroll
  for (int nt = 0; nt < 4; ++nt) {
#pragma unroll
    for (int r = 0; r < 8; ++r) {
      const int m = q0 + wave * 16 + r + 8 * hf;
      const float val = of[nt][r] / lrow[r];
      Aout[((size_t)b * SEQ + m) * D_MODEL + (h * DK + nt * 16 + ln)] = f2bf(val);
    }
  }
}

// ---------------------------------------------------------------------------
extern "C" void kernel_launch(void* const* d_in, const int* in_sizes, int n_in,
                              void* d_out, int out_size, void* d_ws, size_t ws_size,
                              hipStream_t stream) {
  (void)in_sizes; (void)n_in; (void)out_size; (void)ws_size;
  const float* q   = (const float*)d_in[0];
  const float* k   = (const float*)d_in[1];
  const float* v   = (const float*)d_in[2];
  // d_in[3] = bool causal mask (recomputed analytically in-kernel)
  const float* wq  = (const float*)d_in[4];
  const float* bq  = (const float*)d_in[5];
  const float* wk  = (const float*)d_in[6];
  const float* bk  = (const float*)d_in[7];
  const float* wv  = (const float*)d_in[8];
  const float* bv  = (const float*)d_in[9];
  const float* wo  = (const float*)d_in[10];
  const float* bo  = (const float*)d_in[11];

  const size_t NQ = (size_t)4 * NHEAD * SEQ * DK;   // 8 Mi elems (16 MiB bf16)
  bf16_t* Qh = (bf16_t*)d_ws;
  bf16_t* Kh = Qh + NQ;
  bf16_t* Vt = Kh + NQ;
  bf16_t* At = Vt + NQ;

  dim3 gg(8192 / 128, D_MODEL / 128);               // (64, 8)
  mha_gemm<float, 0><<<gg, 256, 0, stream>>>(q, wq, bq, Qh);
  mha_gemm<float, 0><<<gg, 256, 0, stream>>>(k, wk, bk, Kh);
  mha_gemm<float, 1><<<gg, 256, 0, stream>>>(v, wv, bv, Vt);

  mha_flash<<<dim3(SEQ / 64, 4 * NHEAD), 128, 0, stream>>>(Qh, Kh, Vt, At);

  mha_gemm<bf16_t, 2><<<gg, 256, 0, stream>>>(At, wo, bo, (float*)d_out);
}
